// GAT_2997887172896
// MI455X (gfx1250) — matrix operand order repaired and checked by reading, not compiled, not run on hardware
//
#include <hip/hip_runtime.h>

#define H_HEADS 4
#define NEG_SLOPE 0.2f

typedef __bf16 bf16;
typedef __attribute__((ext_vector_type(8)))  __bf16 v8bf;
typedef __attribute__((ext_vector_type(16))) __bf16 v16bf;
typedef __attribute__((ext_vector_type(8)))  float  v8f;

// ---------------- helpers ----------------
__device__ __forceinline__ unsigned f2ord(float f) {
  unsigned u = __float_as_uint(f);
  return (u & 0x80000000u) ? ~u : (u | 0x80000000u);
}
__device__ __forceinline__ float ord2f(unsigned u) {
  return (u & 0x80000000u) ? __uint_as_float(u & 0x7fffffffu)
                           : __uint_as_float(~u);
}
__device__ __forceinline__ void edge_sd(const int* __restrict__ esrc,
                                        const int* __restrict__ edst,
                                        int E, int e, int& s, int& d) {
  if (e < E) { s = esrc[e]; d = edst[e]; }
  else       { s = e - E;   d = s;       }   // appended self-loops
}

// ---------------- generic ----------------
__global__ void k_fill_u32(unsigned* __restrict__ p, unsigned v, int n) {
  int i = blockIdx.x * blockDim.x + threadIdx.x;
  if (i < n) p[i] = v;
}

__global__ void k_cast_bf16(const float* __restrict__ in, bf16* __restrict__ out, int n) {
  int i = blockIdx.x * blockDim.x + threadIdx.x;
  if (i < n) out[i] = (bf16)in[i];
}

__global__ void k_transpose_bf16(const float* __restrict__ W, bf16* __restrict__ Wt,
                                 int K, int Nc) {
  int i = blockIdx.x * blockDim.x + threadIdx.x;
  if (i < K * Nc) {
    int k = i / Nc, n = i % Nc;
    Wt[(size_t)n * K + k] = (bf16)W[i];
  }
}

// ---- WMMA GEMM: C[M][Nc] = A[M][K](bf16) x W; Bt = W^T [Nc][K] bf16.
// Each wave computes a 16 x (NT*16) strip: A fragment reused across NT WMMAs.
template <int NT>
__global__ void k_gemm_wmma_bf16(const bf16* __restrict__ A,
                                 const bf16* __restrict__ Bt,
                                 float* __restrict__ C,
                                 int M, int K, int Nc) {
  int wid  = (blockIdx.x * blockDim.x + threadIdx.x) >> 5;
  int lane = threadIdx.x & 31;
  int tilesN  = Nc >> 4;
  int groupsN = tilesN / NT;
  int nGroups = (M >> 4) * groupsN;
  if (wid >= nGroups) return;              // wave-uniform guard; EXEC all-ones inside
  int tm  = (wid / groupsN) << 4;
  int tn0 = (wid % groupsN) * (NT * 16);
  int l15  = lane & 15;
  int half = lane >> 4;

  // A fragment: row tm+l15; per 32-K step elems 0-7 = K half*8+0..7, elems 8-15 = +16
  const bf16* arow = A + (size_t)(tm + l15) * K + half * 8;
  // B fragment: col (tn + l15) of W == row of Bt; elems 0-7 = K half*8+0..7, elems 8-15 = +16
  const bf16* brow[NT];
  #pragma unroll
  for (int t = 0; t < NT; t++)
    brow[t] = Bt + (size_t)(tn0 + t * 16 + l15) * K + half * 8;

  v8f acc[NT];
  #pragma unroll
  for (int t = 0; t < NT; t++) acc[t] = (v8f){};

  for (int k0 = 0; k0 < K; k0 += 32) {
    v8bf a0 = *(const v8bf*)(arow + k0);
    v8bf a1 = *(const v8bf*)(arow + k0 + 16);
    v16bf a = __builtin_shufflevector(a0, a1, 0,1,2,3,4,5,6,7,8,9,10,11,12,13,14,15);
    #pragma unroll
    for (int t = 0; t < NT; t++) {
      v8bf b0 = *(const v8bf*)(brow[t] + k0);
      v8bf b1 = *(const v8bf*)(brow[t] + k0 + 16);
      v16bf b = __builtin_shufflevector(b0, b1, 0,1,2,3,4,5,6,7,8,9,10,11,12,13,14,15);
      acc[t] = __builtin_amdgcn_wmma_f32_16x16x32_bf16(false, a, false, b,
                                                       (short)0, acc[t], false, false);
    }
  }
  // C/D layout: VGPR r -> row (half*8 + r), col l15
  #pragma unroll
  for (int t = 0; t < NT; t++) {
    float* crow = C + (size_t)(tm + half * 8) * Nc + tn0 + t * 16 + l15;
    #pragma unroll
    for (int r = 0; r < 8; r++) crow[(size_t)r * Nc] = acc[t][r];
  }
}

// ---------------- attention pieces ----------------
__global__ void k_alpha(const float* __restrict__ h, const float* __restrict__ a_src,
                        const float* __restrict__ a_dst, float* __restrict__ as,
                        float* __restrict__ ad, int N, int C) {
  int i = blockIdx.x * blockDim.x + threadIdx.x;
  if (i >= N * H_HEADS) return;
  int n = i / H_HEADS, hd = i % H_HEADS;
  const float* hp = h + (size_t)n * H_HEADS * C + hd * C;
  const float* ps = a_src + hd * C;
  const float* pd = a_dst + hd * C;
  float sa = 0.f, da = 0.f;
  for (int c = 0; c < C; c++) { float v = hp[c]; sa += v * ps[c]; da += v * pd[c]; }
  as[i] = sa; ad[i] = da;
}

__global__ void k_edge_logit_max(const int* __restrict__ esrc, const int* __restrict__ edst,
                                 int E, int Etot,
                                 const float* __restrict__ as, const float* __restrict__ ad,
                                 float* __restrict__ ebuf, unsigned* __restrict__ segmax) {
  int e = blockIdx.x * blockDim.x + threadIdx.x;
  if (e >= Etot) return;
  int s, d; edge_sd(esrc, edst, E, e, s, d);
  #pragma unroll
  for (int hd = 0; hd < H_HEADS; hd++) {
    float v = as[s * H_HEADS + hd] + ad[d * H_HEADS + hd];
    v = v > 0.0f ? v : NEG_SLOPE * v;          // leaky_relu
    ebuf[(size_t)e * H_HEADS + hd] = v;
    atomicMax(&segmax[d * H_HEADS + hd], f2ord(v));
  }
}

__global__ void k_edge_expsum(const int* __restrict__ esrc, const int* __restrict__ edst,
                              int E, int Etot,
                              const unsigned* __restrict__ segmax,
                              float* __restrict__ ebuf, float* __restrict__ segsum) {
  int e = blockIdx.x * blockDim.x + threadIdx.x;
  if (e >= Etot) return;
  int s, d; edge_sd(esrc, edst, E, e, s, d);
  #pragma unroll
  for (int hd = 0; hd < H_HEADS; hd++) {
    float m = ord2f(segmax[d * H_HEADS + hd]);
    float w = expf(ebuf[(size_t)e * H_HEADS + hd] - m);
    ebuf[(size_t)e * H_HEADS + hd] = w;
    atomicAdd(&segsum[d * H_HEADS + hd], w);
  }
}

// one wave per edge: lanes stride across feature channels (L2-resident atomics)
__global__ void k_edge_aggregate(const int* __restrict__ esrc, const int* __restrict__ edst,
                                 int E, int Etot,
                                 const float* __restrict__ hfeat, const float* __restrict__ ebuf,
                                 const float* __restrict__ segsum, float* __restrict__ outacc,
                                 int C) {
  int wid  = (blockIdx.x * blockDim.x + threadIdx.x) >> 5;
  int lane = threadIdx.x & 31;
  if (wid >= Etot) return;
  int s, d; edge_sd(esrc, edst, E, wid, s, d);
  const float* hp = hfeat  + (size_t)s * H_HEADS * C;
  float*       op = outacc + (size_t)d * H_HEADS * C;
  #pragma unroll
  for (int hd = 0; hd < H_HEADS; hd++) {
    float al = ebuf[(size_t)wid * H_HEADS + hd] / (segsum[d * H_HEADS + hd] + 1e-16f);
    int base = hd * C;
    for (int j = lane; j < C; j += 32)
      atomicAdd(&op[base + j], hp[base + j] * al);
  }
}

__global__ void k_bias_elu_cast(const float* __restrict__ acc, const float* __restrict__ b,
                                bf16* __restrict__ xout, int total, int HC) {
  int i = blockIdx.x * blockDim.x + threadIdx.x;
  if (i >= total) return;
  float v = acc[i] + b[i % HC];
  v = v > 0.f ? v : (expf(v) - 1.0f);          // elu
  xout[i] = (bf16)v;
}

__global__ void k_final_mean_bias(const float* __restrict__ acc, const float* __restrict__ b,
                                  float* __restrict__ out, int N, int Cc) {
  int i = blockIdx.x * blockDim.x + threadIdx.x;
  if (i >= N * Cc) return;
  int n = i / Cc, c = i % Cc;
  float s = 0.f;
  #pragma unroll
  for (int hd = 0; hd < H_HEADS; hd++) s += acc[(size_t)n * H_HEADS * Cc + hd * Cc + c];
  out[i] = s * (1.0f / H_HEADS) + b[c];
}

// ---------------- driver ----------------
extern "C" void kernel_launch(void* const* d_in, const int* in_sizes, int n_in,
                              void* d_out, int out_size, void* d_ws, size_t ws_size,
                              hipStream_t stream) {
  const float* x   = (const float*)d_in[0];
  const int*   eix = (const int*)  d_in[1];
  const float* W1  = (const float*)d_in[2];
  const float* as1 = (const float*)d_in[3];
  const float* ad1 = (const float*)d_in[4];
  const float* b1  = (const float*)d_in[5];
  const float* W2  = (const float*)d_in[6];
  const float* as2 = (const float*)d_in[7];
  const float* ad2 = (const float*)d_in[8];
  const float* b2  = (const float*)d_in[9];
  const float* W3  = (const float*)d_in[10];
  const float* as3 = (const float*)d_in[11];
  const float* ad3 = (const float*)d_in[12];
  const float* b3  = (const float*)d_in[13];

  const int FIN = 256;
  int N    = in_sizes[0] / FIN;     // 50000 (divisible by 16)
  int E    = in_sizes[1] / 2;       // 800000
  int Etot = E + N;                 // + self loops
  const int* esrc = eix;
  const int* edst = eix + E;

  size_t off = 0;
  auto take = [&](size_t bytes) -> void* {
    off = (off + 255) & ~(size_t)255;
    void* p = (char*)d_ws + off;
    off += bytes;
    return p;
  };
  bf16*  xb     = (bf16*) take((size_t)N * 256 * sizeof(bf16));
  bf16*  Wt1    = (bf16*) take((size_t)256 * 256 * sizeof(bf16));
  bf16*  Wt2    = (bf16*) take((size_t)256 * 256 * sizeof(bf16));
  bf16*  Wt3    = (bf16*) take((size_t)256 * 160 * sizeof(bf16));
  float* h      = (float*)take((size_t)N * 256 * sizeof(float));
  float* acc    = (float*)take((size_t)N * 256 * sizeof(float));
  float* asb    = (float*)take((size_t)N * 4 * sizeof(float));
  float* adb    = (float*)take((size_t)N * 4 * sizeof(float));
  unsigned* smx = (unsigned*)take((size_t)N * 4 * sizeof(unsigned));
  float* ssum   = (float*)take((size_t)N * 4 * sizeof(float));
  float* ebuf   = (float*)take((size_t)Etot * 4 * sizeof(float));
  (void)ws_size; (void)n_in; (void)out_size;

  const int TPB = 256;
  auto nb = [](long long n, int tpb) { return (int)((n + tpb - 1) / tpb); };

  // weight prep (deterministic, every call)
  k_transpose_bf16<<<nb(256*256, TPB), TPB, 0, stream>>>(W1, Wt1, 256, 256);
  k_transpose_bf16<<<nb(256*256, TPB), TPB, 0, stream>>>(W2, Wt2, 256, 256);
  k_transpose_bf16<<<nb(256*160, TPB), TPB, 0, stream>>>(W3, Wt3, 256, 160);
  k_cast_bf16<<<nb((long long)N*256, TPB), TPB, 0, stream>>>(x, xb, N*256);

  auto layer = [&](const bf16* Wt, int C, const float* asv, const float* adv) {
    int Nc = H_HEADS * C;
    int tilesN = Nc >> 4;
    if (tilesN % 4 == 0) {
      long long waves = (long long)(N / 16) * (tilesN / 4);
      k_gemm_wmma_bf16<4><<<nb(waves*32, TPB), TPB, 0, stream>>>(xb, Wt, h, N, 256, Nc);
    } else {
      long long waves = (long long)(N / 16) * (tilesN / 2);
      k_gemm_wmma_bf16<2><<<nb(waves*32, TPB), TPB, 0, stream>>>(xb, Wt, h, N, 256, Nc);
    }
    k_alpha<<<nb((long long)N*4, TPB), TPB, 0, stream>>>(h, asv, adv, asb, adb, N, C);
    k_fill_u32<<<nb((long long)N*4,  TPB), TPB, 0, stream>>>(smx, 0u, N*4);
    k_fill_u32<<<nb((long long)N*4,  TPB), TPB, 0, stream>>>((unsigned*)ssum, 0u, N*4);
    k_fill_u32<<<nb((long long)N*Nc, TPB), TPB, 0, stream>>>((unsigned*)acc, 0u, N*Nc);
    k_edge_logit_max<<<nb(Etot, TPB), TPB, 0, stream>>>(esrc, edst, E, Etot, asb, adb, ebuf, smx);
    k_edge_expsum  <<<nb(Etot, TPB), TPB, 0, stream>>>(esrc, edst, E, Etot, smx, ebuf, ssum);
    k_edge_aggregate<<<nb((long long)Etot*32, TPB), TPB, 0, stream>>>(esrc, edst, E, Etot,
                                                                      h, ebuf, ssum, acc, C);
  };

  // layer 1: 256 -> 4x64, concat, elu
  layer(Wt1, 64, as1, ad1);
  k_bias_elu_cast<<<nb((long long)N*256, TPB), TPB, 0, stream>>>(acc, b1, xb, N*256, 256);
  // layer 2: 256 -> 4x64, concat, elu
  layer(Wt2, 64, as2, ad2);
  k_bias_elu_cast<<<nb((long long)N*256, TPB), TPB, 0, stream>>>(acc, b2, xb, N*256, 256);
  // layer 3: 256 -> 4x40, head-mean + bias -> d_out
  layer(Wt3, 40, as3, ad3);
  k_final_mean_bias<<<nb((long long)N*40, TPB), TPB, 0, stream>>>(acc, b3, (float*)d_out, N, 40);
}
